// OncogeneHead_39642548142664
// MI455X (gfx1250) — compile-verified, matrix operand order checked by reading, not run on hardware
//
#include <hip/hip_runtime.h>
#include <hip/hip_bf16.h>

// ---------------------------------------------------------------------------
// MI455X / gfx1250 implementation.
// Heavy GEMMs run on V_WMMA_F32_16X16X32_F16 (f16 inputs, f32 accumulate);
// f32->f16 conversion happens in registers during vectorized (b128) staging,
// with a software pipeline: next K-tile loads overlap current tile's WMMAs.
// B is staged TRANSPOSED in LDS so both A and B fragments are two contiguous
// 16-byte runs -> ds_load_b128 pairs, no scalar packing.
// ---------------------------------------------------------------------------

typedef __attribute__((ext_vector_type(16))) _Float16 v16h;
typedef __attribute__((ext_vector_type(8)))  _Float16 v8h;
typedef __attribute__((ext_vector_type(4)))  _Float16 v4h;
typedef __attribute__((ext_vector_type(4)))  float    v4f;
typedef __attribute__((ext_vector_type(8)))  float    v8f;

#define BATCH   8192
#define IN_DIM  2048
#define HID     1024
#define N_ONC   20
#define EMB_D   256
#define CO_HID  512

__device__ __forceinline__ float geluf(float x) {
    // exact GELU: 0.5*x*(1+erf(x/sqrt(2)))
    return 0.5f * x * (1.0f + erff(x * 0.70710678118654752f));
}

template<bool F32>
__device__ __forceinline__ v4h load_cvt4(const void* __restrict__ p, size_t idx) {
    if constexpr (F32) {
        v4f f = *(const v4f*)((const float*)p + idx);
        return __builtin_convertvector(f, v4h);
    } else {
        return *(const v4h*)((const _Float16*)p + idx);
    }
}

// ---------------------------------------------------------------------------
// WMMA GEMM: C(f32, MxN) = A(MxK) @ B(KxN), row-major.
// Block: 128x128, 256 threads = 8 wave32 waves arranged 2x4,
// each wave: 64x32 = 4x2 tiles of 16x16, K-step 32.
// ---------------------------------------------------------------------------
template<bool AF32, bool BF32>
__global__ __launch_bounds__(256)
void gemm_wmma(const void* __restrict__ A, const void* __restrict__ B,
               float* __restrict__ C, int K, int N,
               int lda, int ldb, int ldc) {
    // 40-half (80 B) pitch: b128 fragment reads are bank-conflict free
    __shared__ __align__(16) _Float16 As[128][40];   // A tile: [m][k]
    __shared__ __align__(16) _Float16 Bt[128][40];   // B tile transposed: [n][k]

    const int tid   = threadIdx.x;
    const int lane  = tid & 31;
    const int wave  = tid >> 5;
    const int wrow  = wave >> 2;        // 0..1 -> 64-row strip
    const int wcol  = wave & 3;         // 0..3 -> 32-col strip
    const int lhalf = lane >> 4;        // 0/1 (lane group)
    const int l15   = lane & 15;
    const int m0    = blockIdx.y * 128;
    const int n0    = blockIdx.x * 128;
    const int kb    = lhalf << 3;       // fragment K base: 0 or 8

    const v8f vzero = {0.f,0.f,0.f,0.f,0.f,0.f,0.f,0.f};
    v8f acc[4][2];
#pragma unroll
    for (int ms = 0; ms < 4; ++ms)
#pragma unroll
        for (int ns = 0; ns < 2; ++ns) acc[ms][ns] = vzero;

    v4h aR[4], bR[4];

    auto load_tile = [&](int kt) {
        // A tile 128x32: 1024 groups of 4 contiguous k
#pragma unroll
        for (int p = 0; p < 4; ++p) {
            int g = tid + p * 256;
            int r = g >> 3, c4 = (g & 7) << 2;
            aR[p] = load_cvt4<AF32>(A, (size_t)(m0 + r) * (size_t)lda + (size_t)(kt + c4));
        }
        // B tile 32x128: 1024 groups of 4 contiguous n (coalesced)
#pragma unroll
        for (int p = 0; p < 4; ++p) {
            int g = tid + p * 256;
            int r = g >> 5, c4 = (g & 31) << 2;
            bR[p] = load_cvt4<BF32>(B, (size_t)(kt + r) * (size_t)ldb + (size_t)(n0 + c4));
        }
    };

    auto store_tile = [&]() {
#pragma unroll
        for (int p = 0; p < 4; ++p) {
            int g = tid + p * 256;
            int r = g >> 3, c4 = (g & 7) << 2;
            *(v4h*)&As[r][c4] = aR[p];                  // packed ds_store_b64
        }
#pragma unroll
        for (int p = 0; p < 4; ++p) {
            int g = tid + p * 256;
            int r = g >> 5, c4 = (g & 31) << 2;
#pragma unroll
            for (int u = 0; u < 4; ++u)
                Bt[c4 + u][r] = bR[p][u];               // transpose into LDS
        }
    };

    load_tile(0);
    for (int kt = 0; kt < K; kt += 32) {
        store_tile();
        __syncthreads();
        if (kt + 32 < K) load_tile(kt + 32);            // overlap with WMMAs

        // ---- fragments (ISA 7.12.2 16-bit layouts, wave32) ----
        // lane L, half j:  K = (L>>4)*8 + (j&7) + (j>=8 ? 16 : 0)
        v16h af[4], bf[2];
#pragma unroll
        for (int ms = 0; ms < 4; ++ms) {
            const int row = wrow * 64 + ms * 16 + l15;
            v8h lo = *(const v8h*)&As[row][kb];
            v8h hi = *(const v8h*)&As[row][kb + 16];
            af[ms] = __builtin_shufflevector(lo, hi, 0,1,2,3,4,5,6,7,8,9,10,11,12,13,14,15);
        }
#pragma unroll
        for (int ns = 0; ns < 2; ++ns) {
            const int col = wcol * 32 + ns * 16 + l15;
            v8h lo = *(const v8h*)&Bt[col][kb];
            v8h hi = *(const v8h*)&Bt[col][kb + 16];
            bf[ns] = __builtin_shufflevector(lo, hi, 0,1,2,3,4,5,6,7,8,9,10,11,12,13,14,15);
        }

        // ---- 8 WMMAs: D = A x B + C, f32 accumulate ----
#pragma unroll
        for (int ms = 0; ms < 4; ++ms)
#pragma unroll
            for (int ns = 0; ns < 2; ++ns)
                acc[ms][ns] = __builtin_amdgcn_wmma_f32_16x16x32_f16(
                    false, af[ms], false, bf[ns], (short)0, acc[ms][ns], false, false);
        __syncthreads();
    }

    // ---- epilogue: C/D layout -> VGPR r holds M = r + 8*(lane>=16), N = lane&15
#pragma unroll
    for (int ms = 0; ms < 4; ++ms) {
#pragma unroll
        for (int ns = 0; ns < 2; ++ns) {
            const size_t rbase = (size_t)(m0 + wrow * 64 + ms * 16 + (lhalf << 3));
            const int    col   = n0 + wcol * 32 + ns * 16 + l15;
#pragma unroll
            for (int r = 0; r < 8; ++r)
                C[(rbase + r) * (size_t)ldc + col] = acc[ms][ns][r];
        }
    }
}

// ---------------------------------------------------------------------------
// bias + LayerNorm (population var) + exact GELU, write f16 activations.
// One block (256 threads) per row of 1024.
// ---------------------------------------------------------------------------
__global__ __launch_bounds__(256)
void ln_gelu_kernel(const float* __restrict__ C, const float* __restrict__ bias,
                    const float* __restrict__ g, const float* __restrict__ be,
                    _Float16* __restrict__ O) {
    __shared__ float red[256];
    __shared__ float stats[2];
    const int tid = threadIdx.x;
    const size_t base = (size_t)blockIdx.x * HID;

    float v[4];
    float s = 0.f, ss = 0.f;
#pragma unroll
    for (int u = 0; u < 4; ++u) {
        int c = tid + u * 256;
        float val = C[base + c] + bias[c];
        v[u] = val; s += val; ss += val * val;
    }
    red[tid] = s; __syncthreads();
    for (int o = 128; o > 0; o >>= 1) { if (tid < o) red[tid] += red[tid + o]; __syncthreads(); }
    if (tid == 0) stats[0] = red[0] * (1.0f / (float)HID);
    __syncthreads();
    red[tid] = ss; __syncthreads();
    for (int o = 128; o > 0; o >>= 1) { if (tid < o) red[tid] += red[tid + o]; __syncthreads(); }
    if (tid == 0) {
        float mean = stats[0];
        float var  = red[0] * (1.0f / (float)HID) - mean * mean;
        stats[1] = rsqrtf(var + 1e-5f);
    }
    __syncthreads();
    const float mean = stats[0], rstd = stats[1];
#pragma unroll
    for (int u = 0; u < 4; ++u) {
        int c = tid + u * 256;
        float y = (v[u] - mean) * rstd * g[c] + be[c];
        O[base + c] = (_Float16)geluf(y);
    }
}

// ---------------------------------------------------------------------------
// logits = E @ clf_w + clf_b ; probs = sigmoid(logits).  N=20 is too thin for
// WMMA; one thread per (row, class), E is f16 and L2-hot.
// ---------------------------------------------------------------------------
__global__ void logits_sigmoid_kernel(const _Float16* __restrict__ E,
                                      const float* __restrict__ W,
                                      const float* __restrict__ b,
                                      float* __restrict__ probs) {
    int idx = blockIdx.x * blockDim.x + threadIdx.x;
    if (idx >= BATCH * N_ONC) return;
    int i = idx / N_ONC, j = idx % N_ONC;
    const _Float16* e = E + (size_t)i * HID;
    float a = b[j];
    for (int k = 0; k < HID; ++k) a += (float)e[k] * W[k * N_ONC + j];
    probs[idx] = 1.0f / (1.0f + expf(-a));
}

// embW[j,k] = (emb @ Wb)[j,k] + co_b1[k],  Wb = co_w1[HID:,:]
__global__ void embw_kernel(const float* __restrict__ emb, const float* __restrict__ co_w1,
                            const float* __restrict__ co_b1, float* __restrict__ embW) {
    int idx = blockIdx.x * 256 + threadIdx.x;
    if (idx >= N_ONC * CO_HID) return;
    int j = idx >> 9, k = idx & (CO_HID - 1);
    float a = co_b1[k];
    for (int e = 0; e < EMB_D; ++e)
        a += emb[j * EMB_D + e] * co_w1[(size_t)(HID + e) * CO_HID + k];
    embW[idx] = a;
}

// ---------------------------------------------------------------------------
// s[i,j] = sum_k gelu(P[i,k] + embW[j,k]) * co_w2[k] + co_b2
// co[i,j1,j2] = (j1==j2) ? 0 : s[i, min(j1,j2)]
// One block per row i.
// ---------------------------------------------------------------------------
__global__ __launch_bounds__(256)
void pairwise_kernel(const float* __restrict__ P, const float* __restrict__ embW,
                     const float* __restrict__ w2, const float* __restrict__ b2,
                     float* __restrict__ co) {
    __shared__ float Pr[CO_HID];
    __shared__ float part[N_ONC * 8];
    __shared__ float sS[N_ONC];
    const int tid = threadIdx.x;
    const size_t row = blockIdx.x;

    for (int c = tid; c < CO_HID; c += 256) Pr[c] = P[row * CO_HID + c];
    __syncthreads();

    if (tid < N_ONC * 8) {
        int j = tid >> 3, l = tid & 7;
        const float* ew = embW + j * CO_HID;
        float a = 0.f;
        for (int k = l; k < CO_HID; k += 8)
            a += geluf(Pr[k] + ew[k]) * w2[k];
        part[tid] = a;
    }
    __syncthreads();
    if (tid < N_ONC) {
        float s = b2[0];
#pragma unroll
        for (int u = 0; u < 8; ++u) s += part[tid * 8 + u];
        sS[tid] = s;
    }
    __syncthreads();
    for (int idx = tid; idx < N_ONC * N_ONC; idx += 256) {
        int j1 = idx / N_ONC, j2 = idx % N_ONC;
        co[row * (N_ONC * N_ONC) + idx] = (j1 == j2) ? 0.0f : sS[j1 < j2 ? j1 : j2];
    }
}

// ---------------------------------------------------------------------------
extern "C" void kernel_launch(void* const* d_in, const int* in_sizes, int n_in,
                              void* d_out, int out_size, void* d_ws, size_t ws_size,
                              hipStream_t stream) {
    (void)in_sizes; (void)n_in; (void)out_size; (void)ws_size;
    const float* x     = (const float*)d_in[0];
    const float* W1    = (const float*)d_in[1];
    const float* b1    = (const float*)d_in[2];
    const float* g1    = (const float*)d_in[3];
    const float* be1   = (const float*)d_in[4];
    const float* W2    = (const float*)d_in[5];
    const float* b2    = (const float*)d_in[6];
    const float* g2    = (const float*)d_in[7];
    const float* be2   = (const float*)d_in[8];
    const float* clf_w = (const float*)d_in[9];
    const float* clf_b = (const float*)d_in[10];
    const float* emb   = (const float*)d_in[11];
    const float* co_w1 = (const float*)d_in[12];
    const float* co_b1 = (const float*)d_in[13];
    const float* co_w2 = (const float*)d_in[14];
    const float* co_b2 = (const float*)d_in[15];

    // workspace layout
    char* w = (char*)d_ws;
    const size_t C_BYTES = (size_t)BATCH * HID * sizeof(float);   // 33.5 MB
    const size_t H_BYTES = (size_t)BATCH * HID * 2;               // 16.8 MB (f16)
    float*    Cbuf = (float*)w;                                   // also reused for P
    _Float16* Hbuf = (_Float16*)(w + C_BYTES);
    _Float16* Ebuf = (_Float16*)(w + C_BYTES + H_BYTES);
    float*    embW = (float*)(w + C_BYTES + 2 * H_BYTES);
    float*    Pbuf = Cbuf;  // C consumed before stage-3 GEMM writes P

    float* out_probs = (float*)d_out;                 // 8192 x 20
    float* out_co    = out_probs + BATCH * N_ONC;     // 8192 x 20 x 20

    dim3 blk(256);

    // Stage 1: C = x @ W1 ; H = gelu(LN(C + b1))
    gemm_wmma<true, true><<<dim3(HID / 128, BATCH / 128), blk, 0, stream>>>(
        x, W1, Cbuf, IN_DIM, HID, IN_DIM, HID, HID);
    ln_gelu_kernel<<<BATCH, blk, 0, stream>>>(Cbuf, b1, g1, be1, Hbuf);

    // Stage 2: C = H @ W2 ; E = gelu(LN(C + b2))
    gemm_wmma<false, true><<<dim3(HID / 128, BATCH / 128), blk, 0, stream>>>(
        Hbuf, W2, Cbuf, HID, HID, HID, HID, HID);
    ln_gelu_kernel<<<BATCH, blk, 0, stream>>>(Cbuf, b2, g2, be2, Ebuf);

    // Stage 3a: probs = sigmoid(E @ clf_w + clf_b)
    logits_sigmoid_kernel<<<(BATCH * N_ONC + 255) / 256, blk, 0, stream>>>(
        Ebuf, clf_w, clf_b, out_probs);

    // Stage 3b: P = E @ Wa   (Wa = co_w1[:HID], ldb = CO_HID)
    gemm_wmma<false, true><<<dim3(CO_HID / 128, BATCH / 128), blk, 0, stream>>>(
        Ebuf, co_w1, Pbuf, HID, CO_HID, HID, CO_HID, CO_HID);

    // Stage 4: embW = emb @ Wb + co_b1
    embw_kernel<<<(N_ONC * CO_HID + 255) / 256, blk, 0, stream>>>(emb, co_w1, co_b1, embW);

    // Stage 5: pairwise scores + scatter into co
    pairwise_kernel<<<BATCH, blk, 0, stream>>>(Pbuf, embW, co_w2, co_b2, out_co);
}
